// Seq2SeqAttModel_4105988735220
// MI455X (gfx1250) — matrix operand order repaired and checked by reading, not compiled, not run on hardware
//
#include <hip/hip_runtime.h>

// ---------------- problem dimensions ----------------
constexpr int Bb  = 32;      // batch
constexpr int Ss  = 48;      // src len
constexpr int Tt  = 33;      // trg len
constexpr int Ee  = 620;     // embed
constexpr int Hh  = 1000;    // hidden
constexpr int Aa  = 1000;    // attention
constexpr int Mm  = 500;     // maxout
constexpr int VTt = 30000;   // trg vocab
constexpr int H3  = 3 * Hh;
constexpr int H2  = 2 * Hh;
constexpr int M2  = 2 * Mm;

// ---------------- types ----------------
typedef __attribute__((ext_vector_type(16))) __bf16          v16bf;
typedef __attribute__((ext_vector_type(16))) unsigned short  v16us;
typedef __attribute__((ext_vector_type(8)))  float           v8f;

__device__ __forceinline__ unsigned short f2bf(float f) {
  unsigned u = __builtin_bit_cast(unsigned, f);
  u += 0x7FFFu + ((u >> 16) & 1u);          // round-to-nearest-even
  return (unsigned short)(u >> 16);
}
__device__ __forceinline__ float sigmoidf(float x) { return 1.0f / (1.0f + __expf(-x)); }

// -------- CDNA5 async global->LDS staging (compile-safe detection) --------
#if __has_builtin(__builtin_amdgcn_global_load_async_to_lds_b64)
#define USE_ASYNC_LDS 1
typedef int v2i_gv __attribute__((vector_size(8)));   // generic vector: matches builtin V2i
__device__ __forceinline__ void async_ld64(const void* g, void* l) {
  __builtin_amdgcn_global_load_async_to_lds_b64(
      (__attribute__((address_space(1))) v2i_gv*)g,
      (__attribute__((address_space(3))) v2i_gv*)l, 0, 0);
}
__device__ __forceinline__ void wait_async0() {
#if __has_builtin(__builtin_amdgcn_s_wait_asynccnt)
  __builtin_amdgcn_s_wait_asynccnt(0);
#else
  asm volatile("s_wait_asynccnt 0" ::: "memory");
#endif
}
#endif

// =====================================================================
// bf16 WMMA GEMM:  C[M,N] = A[M,K] * B[N,K]^T (+bias) (+=C if acc)
// A,B: bf16 row-major. C fp32 row-major. Requirements met by caller:
//   M % 32 == 0, K % 4 == 0, lda % 4 == 0, ldb % 4 == 0 (enables b64 I/O).
// Block = 256 threads (8 waves). Tile: 32(M) x 128(N) x 32(K).
// Double-buffered LDS, one barrier per K-tile; tile kt+1 is staged (async
// DMA to LDS when available) while tile kt is computed.
// =====================================================================
__global__ __launch_bounds__(256)
void gemm_bf16_wmma(const unsigned short* __restrict__ A, int lda,
                    const unsigned short* __restrict__ Bm, int ldb,
                    float* __restrict__ C, long ldc,
                    const float* __restrict__ bias,
                    int M, int N, int K, int acc)
{
  // rows padded to 36 ushort = 72 B -> every b64 LDS access stays 8B-aligned
  __shared__ unsigned short lsA[2][32][36];
  __shared__ unsigned short lsB[2][128][36];

  const int tid  = threadIdx.x;
  const int wave = tid >> 5;
  const int lane = tid & 31;
  const int bm   = blockIdx.y * 32;
  const int bn   = blockIdx.x * 128;

  // ---- staging geometry: 1 b64 chunk of A + 4 b64 chunks of B per thread
  const int ar   = tid >> 3;           // A row 0..31
  const int ac4  = (tid & 7) * 4;      // A col group 0..28
  const long arow = (long)(bm + ar) * lda;   // A rows always valid (M % 32 == 0)

  const int ktiles = (K + 31) / 32;

#ifdef USE_ASYNC_LDS
  // issue the 5 async b64 DMAs (or zero-fill OOB-K chunks) for one K-tile
  auto issue = [&](int buf, int k0) {
    {
      int gc = k0 + ac4;
      unsigned short* dst = &lsA[buf][ar][ac4];
      if (gc < K) async_ld64(A + arow + gc, dst);
      else        *(unsigned long long*)dst = 0ull;   // K % 4 == 0: chunk all-OOB
    }
#pragma unroll
    for (int i = 0; i < 4; ++i) {
      int q  = tid * 4 + i;            // chunk id 0..1023
      int r  = q >> 3;                 // B row 0..127
      int c4 = (q & 7) * 4;
      int gr = bn + r;
      int grc = (gr < N) ? gr : (N - 1);   // OOB rows: garbage only feeds unstored C cols
      int gc = k0 + c4;
      unsigned short* dst = &lsB[buf][r][c4];
      if (gc < K) async_ld64(Bm + (long)grc * ldb + gc, dst);
      else        *(unsigned long long*)dst = 0ull;
    }
  };
#else
  unsigned long long ra, rb[4];
  auto loadA = [&](int k0) {
    int gc = k0 + ac4;
    unsigned long long v =
        *(const unsigned long long*)(A + arow + (gc < K ? gc : 0));
    ra = (gc < K) ? v : 0ull;
  };
  auto loadB = [&](int k0) {
#pragma unroll
    for (int i = 0; i < 4; ++i) {
      int q  = tid * 4 + i;
      int r  = q >> 3;
      int c4 = (q & 7) * 4;
      int gr = bn + r;
      int grc = (gr < N) ? gr : (N - 1);
      int gc = k0 + c4;
      unsigned long long v =
          *(const unsigned long long*)(Bm + (long)grc * ldb + (gc < K ? gc : 0));
      rb[i] = (gr < N && gc < K) ? v : 0ull;
    }
  };
  auto stage = [&](int buf) {
    *(unsigned long long*)&lsA[buf][ar][ac4] = ra;
#pragma unroll
    for (int i = 0; i < 4; ++i) {
      int q = tid * 4 + i;
      *(unsigned long long*)&lsB[buf][q >> 3][(q & 7) * 4] = rb[i];
    }
  };
#endif

  v8f acc0 = {};   // rows bm .. bm+15
  v8f acc1 = {};   // rows bm+16 .. bm+31
  const int cl    = lane & 15;
  const int khalf = lane >> 4;

  // ---- prologue: tile 0 resident in buf 0
#ifdef USE_ASYNC_LDS
  issue(0, 0);
  wait_async0();
#else
  loadA(0); loadB(0);
  stage(0);
  if (ktiles > 1) { loadA(32); loadB(32); }
#endif
  __syncthreads();

  for (int kt = 0; kt < ktiles; ++kt) {
    const int buf = kt & 1;

#ifdef USE_ASYNC_LDS
    if (kt + 1 < ktiles) {
      issue((kt + 1) & 1, (kt + 1) * 32);   // DMA overlaps the WMMA work below
      if (kt + 2 < ktiles) {                // gfx1250 global_prefetch_b8 one tile out
        __builtin_prefetch(A + arow + (kt + 2) * 32, 0, 0);
        __builtin_prefetch(Bm + (long)((bn + (tid >> 1)) < N ? (bn + (tid >> 1)) : (N - 1)) * ldb + (kt + 2) * 32, 0, 0);
      }
    }
#endif

    // ---- B fragment (32x16, KxN): lanes 0-15 N=lane K=0..15, lanes 16-31 K=16..31
    v16us bu;
#pragma unroll
    for (int j = 0; j < 16; ++j) bu[j] = lsB[buf][(wave << 4) + cl][khalf * 16 + j];
    v16bf bfr = __builtin_bit_cast(v16bf, bu);

    // ---- A fragments (16x32): element j -> K = khalf*16 + j
    v16us au0, au1;
#pragma unroll
    for (int j = 0; j < 16; ++j) {
      int p = j >> 1, lo = j & 1;
      int k = khalf * 16 + 2 * p + lo;
      au0[j] = lsA[buf][cl][k];
      au1[j] = lsA[buf][16 + cl][k];
    }
    v16bf af0 = __builtin_bit_cast(v16bf, au0);
    v16bf af1 = __builtin_bit_cast(v16bf, au1);

    acc0 = __builtin_amdgcn_wmma_f32_16x16x32_bf16(false, af0, false, bfr, (short)0, acc0, false, false);
    acc1 = __builtin_amdgcn_wmma_f32_16x16x32_bf16(false, af1, false, bfr, (short)0, acc1, false, false);

    if (kt + 1 < ktiles) {
#ifdef USE_ASYNC_LDS
      wait_async0();                   // own DMAs done; barrier makes all waves' visible
#else
      stage((kt + 1) & 1);             // other buffer: safe, protected by prev barrier
      if (kt + 2 < ktiles) {
        const int k0n = (kt + 2) * 32;
        loadA(k0n); loadB(k0n);
        if (kt + 3 < ktiles) {
          __builtin_prefetch(A + arow + (kt + 3) * 32, 0, 0);
          __builtin_prefetch(Bm + (long)((bn + (tid >> 1)) < N ? (bn + (tid >> 1)) : (N - 1)) * ldb + (kt + 3) * 32, 0, 0);
        }
      }
#endif
      __syncthreads();
    }
  }

  // ---- store: VGPR i -> row = bm + (lane>>4)*8 + i, col = bn + wave*16 + (lane&15)
  const int n0 = bn + (wave << 4) + cl;
  if (n0 < N) {
    const float bv = bias ? bias[n0] : 0.0f;
#pragma unroll
    for (int i = 0; i < 8; ++i) {
      int r0 = bm + khalf * 8 + i;
      int r1 = r0 + 16;
      float v0 = acc0[i] + bv;
      float v1 = acc1[i] + bv;
      if (acc) {
        v0 += C[(long)r0 * ldc + n0];
        v1 += C[(long)r1 * ldc + n0];
      }
      C[(long)r0 * ldc + n0] = v0;
      C[(long)r1 * ldc + n0] = v1;
    }
  }
}

// ---------------- fp32 -> bf16 bulk convert ----------------
__global__ void f2bf_kernel(const float* __restrict__ in, unsigned short* __restrict__ out, long n) {
  long i = (long)blockIdx.x * blockDim.x + threadIdx.x;
  long st = (long)gridDim.x * blockDim.x;
  for (; i < n; i += st) out[i] = f2bf(in[i]);
}

// gather embeddings -> bf16, out[(t*B + b)*E + e] = emb[tok[b*tok_ld + t]*E + e]
__global__ void gather_bf16(const int* __restrict__ tok, int tok_ld,
                            const float* __restrict__ emb,
                            unsigned short* __restrict__ out, int TT) {
  long n = (long)TT * Bb * Ee;
  long i = (long)blockIdx.x * blockDim.x + threadIdx.x;
  long st = (long)gridDim.x * blockDim.x;
  for (; i < n; i += st) {
    int e = (int)(i % Ee);
    long r = i / Ee;
    int b = (int)(r % Bb);
    int t = (int)(r / Bb);
    int id = tok[b * tok_ld + t];
    out[i] = f2bf(emb[(long)id * Ee + e]);
  }
}

// zero both hidden states (must re-init every call)
__global__ void init_h(float* hf, float* hb, unsigned short* hfb, unsigned short* hbb) {
  int i = blockIdx.x * blockDim.x + threadIdx.x;
  if (i < Bb * Hh) { hf[i] = 0.f; hb[i] = 0.f; hfb[i] = 0; hbb[i] = 0; }
}

// fused bidirectional encoder GRU elementwise step (PyTorch gate order r,z,n)
__global__ void enc_gru_step(const float* __restrict__ gx_f, const float* __restrict__ gx_b,
                             const float* __restrict__ gh_f, const float* __restrict__ gh_b,
                             float* __restrict__ h_f, float* __restrict__ h_b,
                             unsigned short* __restrict__ h_f_bf, unsigned short* __restrict__ h_b_bf,
                             float* __restrict__ ann, unsigned short* __restrict__ ann_bf, int s) {
  int i = blockIdx.x * blockDim.x + threadIdx.x;
  if (i >= Bb * Hh) return;
  int b = i / Hh, j = i % Hh;
  { // forward: time s -> ann[:, s, 0:H]
    const float* gx = gx_f + ((long)s * Bb + b) * H3;
    const float* gh = gh_f + (long)b * H3;
    float r = sigmoidf(gx[j] + gh[j]);
    float z = sigmoidf(gx[Hh + j] + gh[Hh + j]);
    float n = tanhf(gx[2 * Hh + j] + r * gh[2 * Hh + j]);
    float hn = (1.f - z) * n + z * h_f[i];
    h_f[i] = hn; h_f_bf[i] = f2bf(hn);
    long ai = ((long)b * Ss + s) * H2 + j;
    ann[ai] = hn; ann_bf[ai] = f2bf(hn);
  }
  { // backward: time S-1-s -> ann[:, S-1-s, H:2H]
    int sb = Ss - 1 - s;
    const float* gx = gx_b + ((long)sb * Bb + b) * H3;
    const float* gh = gh_b + (long)b * H3;
    float r = sigmoidf(gx[j] + gh[j]);
    float z = sigmoidf(gx[Hh + j] + gh[Hh + j]);
    float n = tanhf(gx[2 * Hh + j] + r * gh[2 * Hh + j]);
    float hn = (1.f - z) * n + z * h_b[i];
    h_b[i] = hn; h_b_bf[i] = f2bf(hn);
    long ai = ((long)b * Ss + sb) * H2 + Hh + j;
    ann[ai] = hn; ann_bf[ai] = f2bf(hn);
  }
}

// decoder GRU cell elementwise (hidden lives in h / hbf)
__global__ void gru_cell_step(const float* __restrict__ gx, const float* __restrict__ gh,
                              float* __restrict__ h, unsigned short* __restrict__ hbf) {
  int i = blockIdx.x * blockDim.x + threadIdx.x;
  if (i >= Bb * Hh) return;
  int b = i / Hh, j = i % Hh;
  const float* gxb = gx + (long)b * H3;
  const float* ghb = gh + (long)b * H3;
  float r = sigmoidf(gxb[j] + ghb[j]);
  float z = sigmoidf(gxb[Hh + j] + ghb[Hh + j]);
  float n = tanhf(gxb[2 * Hh + j] + r * ghb[2 * Hh + j]);
  float hn = (1.f - z) * n + z * h[i];
  h[i] = hn; hbf[i] = f2bf(hn);
}

// scores[b,s] = sum_a va[a] * tanh(wa_h[b,a] + ua_ann[b,s,a]); one wave per (b,s)
__global__ __launch_bounds__(256)
void attn_scores(const float* __restrict__ wa_h, const float* __restrict__ ua_ann,
                 const float* __restrict__ va, float* __restrict__ scores) {
  int wid = blockIdx.x * 8 + (threadIdx.x >> 5);
  int lane = threadIdx.x & 31;
  if (wid >= Bb * Ss) return;
  int b = wid / Ss, s = wid % Ss;
  const float* wh = wa_h + (long)b * Aa;
  const float* ua = ua_ann + ((long)b * Ss + s) * Aa;
  float sum = 0.f;
  for (int a = lane; a < Aa; a += 32) sum += va[a] * tanhf(wh[a] + ua[a]);
#pragma unroll
  for (int off = 16; off > 0; off >>= 1) sum += __shfl_xor(sum, off, 32);
  if (lane == 0) scores[wid] = sum;
}

// softmax over S per batch row; lane b handles row b
__global__ void attn_softmax(const float* __restrict__ scores, float* __restrict__ alpha) {
  int b = threadIdx.x;
  if (b >= Bb) return;
  const float* sr = scores + (long)b * Ss;
  float m = -3.0e38f;
  for (int s = 0; s < Ss; ++s) m = fmaxf(m, sr[s]);
  float sum = 0.f;
  for (int s = 0; s < Ss; ++s) sum += __expf(sr[s] - m);
  float inv = 1.f / sum;
  for (int s = 0; s < Ss; ++s) alpha[b * Ss + s] = __expf(sr[s] - m) * inv;
}

// context[b,d] = sum_s alpha[b,s] * ann[b,s,d]
__global__ void attn_context(const float* __restrict__ alpha, const float* __restrict__ ann,
                             float* __restrict__ ctx, unsigned short* __restrict__ ctx_bf) {
  int i = blockIdx.x * blockDim.x + threadIdx.x;
  if (i >= Bb * H2) return;
  int b = i / H2, d = i % H2;
  float acc = 0.f;
  for (int s = 0; s < Ss; ++s) acc += alpha[b * Ss + s] * ann[((long)b * Ss + s) * H2 + d];
  ctx[i] = acc; ctx_bf[i] = f2bf(acc);
}

// maxout over adjacent pairs; ve_rows already includes e_t@Vo^T + bvo
__global__ void maxout_fin(const float* __restrict__ t2m, const float* __restrict__ ve_rows,
                           unsigned short* __restrict__ tmax) {
  int i = blockIdx.x * blockDim.x + threadIdx.x;
  if (i >= Bb * Mm) return;
  int b = i / Mm, m = i % Mm;
  const float* tb = t2m + (long)b * M2;
  const float* vb = ve_rows + (long)b * M2;
  float a0 = tb[2 * m] + vb[2 * m];
  float a1 = tb[2 * m + 1] + vb[2 * m + 1];
  tmax[i] = f2bf(fmaxf(a0, a1));
}

// =====================================================================
extern "C" void kernel_launch(void* const* d_in, const int* in_sizes, int n_in,
                              void* d_out, int out_size, void* d_ws, size_t ws_size,
                              hipStream_t stream) {
  (void)in_sizes; (void)n_in; (void)out_size; (void)ws_size;
  // ---- inputs (setup_inputs order) ----
  const int*   src     = (const int*)d_in[0];
  const int*   trg     = (const int*)d_in[2];
  const float* src_emb = (const float*)d_in[4];
  const float* trg_emb = (const float*)d_in[5];
  const float* Wih_f = (const float*)d_in[6],  *Whh_f = (const float*)d_in[7];
  const float* bih_f = (const float*)d_in[8],  *bhh_f = (const float*)d_in[9];
  const float* Wih_b = (const float*)d_in[10], *Whh_b = (const float*)d_in[11];
  const float* bih_b = (const float*)d_in[12], *bhh_b = (const float*)d_in[13];
  const float* Wih_d = (const float*)d_in[14], *Whh_d = (const float*)d_in[15];
  const float* bih_d = (const float*)d_in[16], *bhh_d = (const float*)d_in[17];
  const float* Wa  = (const float*)d_in[18], *ba  = (const float*)d_in[19];
  const float* Ua  = (const float*)d_in[20], *bua = (const float*)d_in[21];
  const float* va  = (const float*)d_in[22];
  const float* Uo  = (const float*)d_in[23], *buo = (const float*)d_in[24];
  const float* Vo  = (const float*)d_in[25], *bvo = (const float*)d_in[26];
  const float* Co  = (const float*)d_in[27], *bco = (const float*)d_in[28];
  const float* Wo  = (const float*)d_in[29], *bwo = (const float*)d_in[30];
  float* out = (float*)d_out;

  // ---- workspace carve-out (~155 MB total) ----
  char* wsb = (char*)d_ws;
  size_t off = 0;
  auto alloc = [&](size_t bytes) -> void* {
    off = (off + 255) & ~(size_t)255;
    void* p = wsb + off; off += bytes; return p;
  };
  auto abf = [&](long n) { return (unsigned short*)alloc((size_t)n * 2 + 128); };
  auto af32 = [&](long n) { return (float*)alloc((size_t)n * 4); };

  // bf16 weight copies
  unsigned short* wih_f_bf = abf((long)H3 * Ee);
  unsigned short* whh_f_bf = abf((long)H3 * Hh);
  unsigned short* wih_b_bf = abf((long)H3 * Ee);
  unsigned short* whh_b_bf = abf((long)H3 * Hh);
  unsigned short* wih_d_bf = abf((long)H3 * (Ee + H2));
  unsigned short* whh_d_bf = abf((long)H3 * Hh);
  unsigned short* wa_bf    = abf((long)Aa * Hh);
  unsigned short* ua_bf    = abf((long)Aa * H2);
  unsigned short* uo_bf    = abf((long)M2 * Hh);
  unsigned short* vo_bf    = abf((long)M2 * Ee);
  unsigned short* co_bf    = abf((long)M2 * H2);
  unsigned short* wo_bf    = abf((long)VTt * Mm);
  // activations
  unsigned short* x_src  = abf((long)Ss * Bb * Ee);        // [S,B,E] bf16
  unsigned short* e_all  = abf((long)(Tt - 1) * Bb * Ee);  // [T-1,B,E] bf16
  float* gx_f   = af32((long)Ss * Bb * H3);
  float* gx_b   = af32((long)Ss * Bb * H3);
  float* ann    = af32((long)Bb * Ss * H2);
  unsigned short* ann_bf = abf((long)Bb * Ss * H2);
  float* ua_ann = af32((long)Bb * Ss * Aa);
  float* ve     = af32((long)(Tt - 1) * Bb * M2);          // e@Vo^T + bvo
  float* h_f    = af32((long)Bb * Hh);
  float* h_b    = af32((long)Bb * Hh);                     // doubles as decoder hidden
  unsigned short* h_f_bf = abf((long)Bb * Hh);
  unsigned short* h_b_bf = abf((long)Bb * Hh);
  float* gh_f = af32((long)Bb * H3);
  float* gh_b = af32((long)Bb * H3);
  float* wa_h = af32((long)Bb * Aa);
  float* scores = af32((long)Bb * Ss);
  float* alpha  = af32((long)Bb * Ss);
  float* ctx    = af32((long)Bb * H2);
  unsigned short* ctx_bf = abf((long)Bb * H2);
  float* t2m  = af32((long)Bb * M2);
  unsigned short* tmax_bf = abf((long)Bb * Mm);
  float* gx_d = af32((long)Bb * H3);
  float* gh_d = af32((long)Bb * H3);

  auto cvt = [&](const float* s, unsigned short* d, long n) {
    int g = (int)((n + 255) / 256); if (g > 4096) g = 4096;
    f2bf_kernel<<<g, 256, 0, stream>>>(s, d, n);
  };
  auto gemm = [&](const unsigned short* A, int lda, const unsigned short* Bm, int ldb,
                  float* C, long ldc, const float* bias, int M, int N, int K, int acc) {
    dim3 grid((N + 127) / 128, (M + 31) / 32);
    gemm_bf16_wmma<<<grid, 256, 0, stream>>>(A, lda, Bm, ldb, C, ldc, bias, M, N, K, acc);
  };

  // ---- phase 0: weight conversion ----
  cvt(Wih_f, wih_f_bf, (long)H3 * Ee);   cvt(Whh_f, whh_f_bf, (long)H3 * Hh);
  cvt(Wih_b, wih_b_bf, (long)H3 * Ee);   cvt(Whh_b, whh_b_bf, (long)H3 * Hh);
  cvt(Wih_d, wih_d_bf, (long)H3 * (Ee + H2));
  cvt(Whh_d, whh_d_bf, (long)H3 * Hh);
  cvt(Wa, wa_bf, (long)Aa * Hh);         cvt(Ua, ua_bf, (long)Aa * H2);
  cvt(Uo, uo_bf, (long)M2 * Hh);         cvt(Vo, vo_bf, (long)M2 * Ee);
  cvt(Co, co_bf, (long)M2 * H2);         cvt(Wo, wo_bf, (long)VTt * Mm);

  // ---- phase 1: embedding gathers ----
  gather_bf16<<<4096, 256, 0, stream>>>(src, Ss, src_emb, x_src, Ss);
  gather_bf16<<<2048, 256, 0, stream>>>(trg, Tt, trg_emb, e_all, Tt - 1);

  // ---- phase 2: batched input projections ----
  // gx_f/gx_b: [S*B, 3H] = x @ Wih^T + bih  (M=1536)
  gemm(x_src, Ee, wih_f_bf, Ee, gx_f, H3, bih_f, Ss * Bb, H3, Ee, 0);
  gemm(x_src, Ee, wih_b_bf, Ee, gx_b, H3, bih_b, Ss * Bb, H3, Ee, 0);
  // ve: [(T-1)*B, 2M] = e @ Vo^T + bvo  (M=1024)
  gemm(e_all, Ee, vo_bf, Ee, ve, M2, bvo, (Tt - 1) * Bb, M2, Ee, 0);

  // ---- phase 3: bidirectional encoder scan ----
  init_h<<<(Bb * Hh + 255) / 256, 256, 0, stream>>>(h_f, h_b, h_f_bf, h_b_bf);
  for (int s = 0; s < Ss; ++s) {
    gemm(h_f_bf, Hh, whh_f_bf, Hh, gh_f, H3, bhh_f, Bb, H3, Hh, 0);
    gemm(h_b_bf, Hh, whh_b_bf, Hh, gh_b, H3, bhh_b, Bb, H3, Hh, 0);
    enc_gru_step<<<(Bb * Hh + 255) / 256, 256, 0, stream>>>(
        gx_f, gx_b, gh_f, gh_b, h_f, h_b, h_f_bf, h_b_bf, ann, ann_bf, s);
  }

  // ---- phase 4: ua_ann = ann @ Ua^T + bua  (M=1536) ----
  gemm(ann_bf, H2, ua_bf, H2, ua_ann, Aa, bua, Bb * Ss, Aa, H2, 0);

  // ---- phase 5: decoder scan (hidden = h_b == encoder backward final state) ----
  for (int t = 0; t < Tt - 1; ++t) {
    // attention
    gemm(h_b_bf, Hh, wa_bf, Hh, wa_h, Aa, ba, Bb, Aa, Hh, 0);
    attn_scores<<<(Bb * Ss + 7) / 8, 256, 0, stream>>>(wa_h, ua_ann, va, scores);
    attn_softmax<<<1, 32, 0, stream>>>(scores, alpha);
    attn_context<<<(Bb * H2 + 255) / 256, 256, 0, stream>>>(alpha, ann, ctx, ctx_bf);
    // maxout readout: t2m = h@Uo^T + buo + ctx@Co^T + bco ; (+ ve row in finalize)
    gemm(h_b_bf, Hh, uo_bf, Hh, t2m, M2, buo, Bb, M2, Hh, 0);
    gemm(ctx_bf, H2, co_bf, H2, t2m, M2, bco, Bb, M2, H2, 1);
    maxout_fin<<<(Bb * Mm + 255) / 256, 256, 0, stream>>>(t2m, ve + (long)t * Bb * M2, tmax_bf);
    // logits -> out[b, t, :]  (row stride (T-1)*VT, base offset t*VT)
    gemm(tmax_bf, Mm, wo_bf, Mm, out + (long)t * VTt, (long)(Tt - 1) * VTt,
         bwo, Bb, VTt, Mm, 0);
    // decoder GRU cell: input = concat(e_t, ctx) -> split Wih_d by columns
    gemm(e_all + (long)t * Bb * Ee, Ee, wih_d_bf, Ee + H2, gx_d, H3, bih_d, Bb, H3, Ee, 0);
    gemm(ctx_bf, H2, wih_d_bf + Ee, Ee + H2, gx_d, H3, nullptr, Bb, H3, H2, 1);
    gemm(h_b_bf, Hh, whh_d_bf, Hh, gh_d, H3, bhh_d, Bb, H3, Hh, 0);
    gru_cell_step<<<(Bb * Hh + 255) / 256, 256, 0, stream>>>(gx_d, gh_d, h_b, h_b_bf);
  }
}
